// CacheEvictionTransformer_5205500363149
// MI455X (gfx1250) — matrix operand order repaired and checked by reading, not compiled, not run on hardware
//
#include <hip/hip_runtime.h>
#include <hip/hip_bf16.h>

// ---------------------------------------------------------------------------
// Problem dimensions (fixed by the reference)
// ---------------------------------------------------------------------------
#define B_   16
#define K_   64
#define W_   2048
#define D_   512
#define DH_  256
#define DFF_ 2048
#define NL_  4
#define L_   (K_ + W_)    // 2112
#define BL_  (B_ * L_)    // 33792

typedef __attribute__((ext_vector_type(16))) _Float16 v16h;
typedef __attribute__((ext_vector_type(8)))  _Float16 v8h;
typedef __attribute__((ext_vector_type(8)))  float    v8f;

// ---------------------------------------------------------------------------
// Reduction helpers (wave32)
// ---------------------------------------------------------------------------
__device__ inline float block_sum(float v, float* sm) {
  const int lane = threadIdx.x & 31;
  const int w    = threadIdx.x >> 5;
  const int nw   = blockDim.x >> 5;
#pragma unroll
  for (int o = 16; o; o >>= 1) v += __shfl_xor(v, o, 32);
  if (lane == 0) sm[w] = v;
  __syncthreads();
  float t = (threadIdx.x < nw) ? sm[threadIdx.x] : 0.f;
  if (w == 0) {
#pragma unroll
    for (int o = 16; o; o >>= 1) t += __shfl_xor(t, o, 32);
    if (lane == 0) sm[0] = t;
  }
  __syncthreads();
  float r = sm[0];
  __syncthreads();
  return r;
}

__device__ inline float block_max(float v, float* sm) {
  const int lane = threadIdx.x & 31;
  const int w    = threadIdx.x >> 5;
  const int nw   = blockDim.x >> 5;
#pragma unroll
  for (int o = 16; o; o >>= 1) v = fmaxf(v, __shfl_xor(v, o, 32));
  if (lane == 0) sm[w] = v;
  __syncthreads();
  float t = (threadIdx.x < nw) ? sm[threadIdx.x] : -3.4e38f;
  if (w == 0) {
#pragma unroll
    for (int o = 16; o; o >>= 1) t = fmaxf(t, __shfl_xor(t, o, 32));
    if (lane == 0) sm[0] = t;
  }
  __syncthreads();
  float r = sm[0];
  __syncthreads();
  return r;
}

// ---------------------------------------------------------------------------
// f16 fragment loaders (CDNA5 wave32 WMMA layouts), dense 16B vector loads.
//
// A (16x32): lane m = lane&15; K-half kh = (lane>>4)*8;
//   f[0..7]  = A[m][k0+kh .. k0+kh+7]       (one b128)
//   f[8..15] = A[m][k0+16+kh .. k0+16+kh+7] (one b128)
// B (32x16): lane n = lane&15; f[0..7] = B[k0+kh+i][n], f[8..15] = B[k0+16+kh+i][n] (strided)
// ---------------------------------------------------------------------------
__device__ inline v16h ld_a16(const _Float16* __restrict__ A, int lda,
                              int m0, int k0, int lane) {
  const _Float16* p = A + (long long)(m0 + (lane & 15)) * lda + k0 + ((lane >> 4) * 8);
  v8h lo = *(const v8h*)(p);
  v8h hi = *(const v8h*)(p + 16);
  return __builtin_shufflevector(lo, hi, 0, 1, 2, 3, 4, 5, 6, 7,
                                         8, 9, 10, 11, 12, 13, 14, 15);
}

__device__ inline v16h ld_b16(const _Float16* __restrict__ Bm, int ldb,
                              int k0, int n0, int lane) {
  const _Float16* p = Bm + (long long)(k0 + ((lane >> 4) * 8)) * ldb + n0 + (lane & 15);
  v16h f;
#pragma unroll
  for (int i = 0; i < 8; ++i) {
    f[i]     = p[(long long)i * ldb];
    f[8 + i] = p[(long long)(16 + i) * ldb];
  }
  return f;
}

// ---------------------------------------------------------------------------
// Batched GEMM, f16 operands, f32 WMMA accumulation.
//   OUT = 0 : f32 C (+bias, +f32 residual, opt ReLU)
//   OUT = 1 : f16 C (+bias, opt ReLU)
//   OUT = 2 : f16 C stored TRANSPOSED (C_T[n][m]), contiguous v8h stores
// Block = 128 threads = 4 waves; wave computes 64x64 (4x4 WMMA tiles):
// 8 b128 A loads + 64 strided f16 B loads per 16 WMMA per k-step.
// M multiple of 64 at wave granularity; N multiple of 64; Kd multiple of 32.
// ---------------------------------------------------------------------------
template <int OUT, bool RELU>
__global__ __launch_bounds__(128)
void gemm16(const _Float16* __restrict__ A, int lda, long long sA,
            const _Float16* __restrict__ Bm, int ldb, long long sB,
            void* __restrict__ Cv, int ldc, long long sC,
            const float* __restrict__ bias,
            const float* __restrict__ res, long long sR,
            int M, int N, int Kd) {
  const int lane = threadIdx.x & 31;
  const int wave = threadIdx.x >> 5;
  const int bz   = blockIdx.z;
  A  += (long long)bz * sA;
  Bm += (long long)bz * sB;
  if (res) res += (long long)bz * sR;

  const int mbase = blockIdx.y * 256 + wave * 64;
  if (mbase >= M) return;
  const int nbase = blockIdx.x * 64;

  const long long laneB = (long long)lane * ldb + nbase;  // k-invariant part

  v8f acc[4][4] = {};
  for (int k0 = 0; k0 < Kd; k0 += 32) {
    // Branch-free L2 prefetch of next k-step's B rows (global_prefetch_b8).
    int kp = k0 + 32;
    kp = (kp < Kd) ? kp : 0;
    __builtin_prefetch(Bm + (long long)kp * ldb + laneB, 0, 3);

    v16h bf[4];
#pragma unroll
    for (int ni = 0; ni < 4; ++ni)
      bf[ni] = ld_b16(Bm, ldb, k0, nbase + ni * 16, lane);
#pragma unroll
    for (int mi = 0; mi < 4; ++mi) {
      v16h af = ld_a16(A, lda, mbase + mi * 16, k0, lane);
#pragma unroll
      for (int ni = 0; ni < 4; ++ni)
        acc[mi][ni] = __builtin_amdgcn_wmma_f32_16x16x32_f16(
            false, af, false, bf[ni], (short)0, acc[mi][ni], false, false);
    }
  }

  // C/D layout: VGPR r -> row = r + 8*lane[4]; col = lane&15
  const int rhalf = (lane >> 4) * 8;
  const int cn    = lane & 15;
#pragma unroll
  for (int mi = 0; mi < 4; ++mi) {
#pragma unroll
    for (int ni = 0; ni < 4; ++ni) {
      const int col   = nbase + ni * 16 + cn;
      const int rbase = mbase + mi * 16 + rhalf;
      const float bv  = bias ? bias[col] : 0.f;
      if (OUT == 0) {
        float* C = (float*)Cv + (long long)bz * sC;
#pragma unroll
        for (int r = 0; r < 8; ++r) {
          float v = acc[mi][ni][r] + bv;
          if (res) v += res[(long long)(rbase + r) * ldc + col];
          if (RELU) v = fmaxf(v, 0.f);
          C[(long long)(rbase + r) * ldc + col] = v;
        }
      } else if (OUT == 1) {
        _Float16* C = (_Float16*)Cv + (long long)bz * sC;
#pragma unroll
        for (int r = 0; r < 8; ++r) {
          float v = acc[mi][ni][r] + bv;
          if (RELU) v = fmaxf(v, 0.f);
          C[(long long)(rbase + r) * ldc + col] = (_Float16)v;
        }
      } else {  // transposed f16: C_T[col][rbase..rbase+7], one 16B store
        _Float16* C = (_Float16*)Cv + (long long)bz * sC;
        v8h hv;
#pragma unroll
        for (int r = 0; r < 8; ++r) hv[r] = (_Float16)(acc[mi][ni][r] + bv);
        *(v8h*)(C + (long long)col * ldc + rbase) = hv;
      }
    }
  }
}

// ---------------------------------------------------------------------------
// f32 -> f16 conversion (for weights, once per launch)
// ---------------------------------------------------------------------------
__global__ __launch_bounds__(256)
void cvt_f32_f16(const float* __restrict__ x, _Float16* __restrict__ y, long long n) {
  long long i = (long long)blockIdx.x * blockDim.x + threadIdx.x;
  if (i < n) y[i] = (_Float16)x[i];
}

// ---------------------------------------------------------------------------
// Embedding: h[b,l,:] = item[tok] + pos + segment   (f32 master)
// ---------------------------------------------------------------------------
__global__ __launch_bounds__(256)
void embed_kernel(const int* __restrict__ cache, const int* __restrict__ seq,
                  const float* __restrict__ item, const float* __restrict__ cpos,
                  const float* __restrict__ spos, const float* __restrict__ seg,
                  float* __restrict__ h) {
  const long long row = blockIdx.x;
  const int b = (int)(row / L_);
  const int l = (int)(row % L_);
  int tok;
  const float* pos;
  const float* sg;
  if (l < K_) {
    tok = cache[b * K_ + l];
    pos = cpos + (long long)l * D_;
    sg  = seg;
  } else {
    tok = seq[b * W_ + (l - K_)];
    pos = spos + (long long)(l - K_) * D_;
    sg  = seg + D_;
  }
  const float* it = item + (long long)tok * D_;
  float* out = h + row * D_;
  for (int i = threadIdx.x; i < D_; i += blockDim.x)
    out[i] = it[i] + pos[i] + sg[i];
}

// ---------------------------------------------------------------------------
// LayerNorm over last dim, f32 in, f16 out (feeds WMMA GEMMs directly)
// ---------------------------------------------------------------------------
__global__ __launch_bounds__(256)
void ln16_kernel(const float* __restrict__ x, const float* __restrict__ g,
                 const float* __restrict__ bta, _Float16* __restrict__ y, int Dn) {
  __shared__ float sm[8];
  const long long row = blockIdx.x;
  const float* xr = x + row * Dn;
  _Float16* yr = y + row * Dn;
  float s = 0.f;
  for (int i = threadIdx.x; i < Dn; i += blockDim.x) s += xr[i];
  const float mu = block_sum(s, sm) / (float)Dn;
  float vsum = 0.f;
  for (int i = threadIdx.x; i < Dn; i += blockDim.x) {
    float d = xr[i] - mu;
    vsum += d * d;
  }
  const float var = block_sum(vsum, sm) / (float)Dn;
  const float inv = rsqrtf(var + 1e-5f);
  for (int i = threadIdx.x; i < Dn; i += blockDim.x)
    yr[i] = (_Float16)((xr[i] - mu) * inv * g[i] + bta[i]);
}

// ---------------------------------------------------------------------------
// In-place f16 softmax with fused scale + causal window (keys <= l-K).
// Masked columns written as exact 0 so the context GEMM is exact.
// ---------------------------------------------------------------------------
__global__ __launch_bounds__(256)
void softmax16_kernel(_Float16* __restrict__ S, int ncols, float scale,
                      int Lrows, int Koff, int causal) {
  __shared__ float sm[8];
  const long long row = blockIdx.x;
  _Float16* r = S + row * (long long)ncols;
  int limit = ncols;
  if (causal) {
    const int l = (int)(row % Lrows);
    if (l >= Koff) {
      int lim = l - Koff + 1;
      limit = lim < ncols ? lim : ncols;
    }
  }
  float mx = -3.4e38f;
  for (int i = threadIdx.x; i < limit; i += blockDim.x)
    mx = fmaxf(mx, (float)r[i] * scale);
  mx = block_max(mx, sm);

  float ev[8];
  float sum = 0.f;
  int j = 0;
  for (int i = threadIdx.x; i < ncols; i += blockDim.x, ++j) {
    float e = (i < limit) ? __expf((float)r[i] * scale - mx) : 0.f;
    ev[j] = e;
    sum += e;
  }
  sum = block_sum(sum, sm);
  const float inv = 1.f / sum;
  j = 0;
  for (int i = threadIdx.x; i < ncols; i += blockDim.x, ++j)
    r[i] = (_Float16)(ev[j] * inv);
}

// ---------------------------------------------------------------------------
// Final head: logits[b,k] = dot(hn16[b,k,:], w_ev) + b_ev
// ---------------------------------------------------------------------------
__global__ __launch_bounds__(256)
void logits_kernel(const _Float16* __restrict__ hn, const float* __restrict__ w_ev,
                   const float* __restrict__ b_ev, float* __restrict__ out) {
  __shared__ float sm[8];
  const int idx = blockIdx.x;  // b*K_ + k
  const int b = idx / K_;
  const int k = idx % K_;
  const _Float16* row = hn + ((long long)b * L_ + k) * D_;
  float s = 0.f;
  for (int i = threadIdx.x; i < D_; i += blockDim.x) s += (float)row[i] * w_ev[i];
  s = block_sum(s, sm);
  if (threadIdx.x == 0) out[idx] = s + b_ev[0];
}

// ---------------------------------------------------------------------------
// Host orchestration
// ---------------------------------------------------------------------------
extern "C" void kernel_launch(void* const* d_in, const int* in_sizes, int n_in,
                              void* d_out, int out_size, void* d_ws, size_t ws_size,
                              hipStream_t stream) {
  const int*   cache = (const int*)d_in[0];
  const int*   seq   = (const int*)d_in[1];
  const float* item  = (const float*)d_in[2];
  const float* cpos  = (const float*)d_in[3];
  const float* spos  = (const float*)d_in[4];
  const float* seg   = (const float*)d_in[5];
  const float* wq_s  = (const float*)d_in[6];
  const float* wk_s  = (const float*)d_in[7];
  const float* wv_s  = (const float*)d_in[8];
  const float* wq_c  = (const float*)d_in[9];
  const float* wk_c  = (const float*)d_in[10];
  const float* wv_c  = (const float*)d_in[11];
  const float* w_out = (const float*)d_in[12];
  const float* b_out = (const float*)d_in[13];
  const float* ln1_g = (const float*)d_in[14];
  const float* ln1_b = (const float*)d_in[15];
  const float* ln2_g = (const float*)d_in[16];
  const float* ln2_b = (const float*)d_in[17];
  const float* w_ff1 = (const float*)d_in[18];
  const float* b_ff1 = (const float*)d_in[19];
  const float* w_ff2 = (const float*)d_in[20];
  const float* b_ff2 = (const float*)d_in[21];
  const float* fin_g = (const float*)d_in[22];
  const float* fin_b = (const float*)d_in[23];
  const float* w_ev  = (const float*)d_in[24];
  const float* b_ev  = (const float*)d_in[25];
  (void)in_sizes; (void)n_in; (void)out_size; (void)ws_size;

  // ------------------------------ workspace ------------------------------
  char* ws = (char*)d_ws;
  size_t off = 0;
  auto alloc = [&](size_t bytes) -> void* {
    void* p = ws + off;
    off = (off + bytes + 255) & ~(size_t)255;
    return p;
  };
  float*     h     = (float*)alloc((size_t)BL_ * D_ * 4);
  _Float16*  hn16  = (_Float16*)alloc((size_t)BL_ * D_ * 2);
  _Float16*  qa16  = (_Float16*)alloc((size_t)BL_ * DH_ * 2);
  _Float16*  qb16  = (_Float16*)alloc((size_t)BL_ * DH_ * 2);
  _Float16*  kaT   = (_Float16*)alloc((size_t)B_ * DH_ * W_ * 2);  // DH x W per batch
  _Float16*  va16  = (_Float16*)alloc((size_t)B_ * W_ * DH_ * 2);
  _Float16*  kbT   = (_Float16*)alloc((size_t)B_ * DH_ * K_ * 2);  // DH x K per batch
  _Float16*  vb16  = (_Float16*)alloc((size_t)B_ * K_ * DH_ * 2);
  _Float16*  at16  = (_Float16*)alloc((size_t)BL_ * D_ * 2);       // attn concat
  _Float16*  sc16  = (_Float16*)alloc((size_t)BL_ * W_ * 2);       // scores_a / FFN hidden
  _Float16*  scb16 = (_Float16*)alloc((size_t)BL_ * K_ * 2);       // scores_b
  // f16 weights
  _Float16* wq_s16  = (_Float16*)alloc((size_t)NL_ * D_ * DH_ * 2);
  _Float16* wk_s16  = (_Float16*)alloc((size_t)NL_ * D_ * DH_ * 2);
  _Float16* wv_s16  = (_Float16*)alloc((size_t)NL_ * D_ * DH_ * 2);
  _Float16* wq_c16  = (_Float16*)alloc((size_t)NL_ * D_ * DH_ * 2);
  _Float16* wk_c16  = (_Float16*)alloc((size_t)NL_ * D_ * DH_ * 2);
  _Float16* wv_c16  = (_Float16*)alloc((size_t)NL_ * D_ * DH_ * 2);
  _Float16* w_out16 = (_Float16*)alloc((size_t)NL_ * D_ * D_ * 2);
  _Float16* w_ff116 = (_Float16*)alloc((size_t)NL_ * D_ * DFF_ * 2);
  _Float16* w_ff216 = (_Float16*)alloc((size_t)NL_ * DFF_ * D_ * 2);

  auto cvt = [&](const float* src, _Float16* dst, long long n) {
    cvt_f32_f16<<<(unsigned)((n + 255) / 256), 256, 0, stream>>>(src, dst, n);
  };
  cvt(wq_s, wq_s16, (long long)NL_ * D_ * DH_);
  cvt(wk_s, wk_s16, (long long)NL_ * D_ * DH_);
  cvt(wv_s, wv_s16, (long long)NL_ * D_ * DH_);
  cvt(wq_c, wq_c16, (long long)NL_ * D_ * DH_);
  cvt(wk_c, wk_c16, (long long)NL_ * D_ * DH_);
  cvt(wv_c, wv_c16, (long long)NL_ * D_ * DH_);
  cvt(w_out, w_out16, (long long)NL_ * D_ * D_);
  cvt(w_ff1, w_ff116, (long long)NL_ * D_ * DFF_);
  cvt(w_ff2, w_ff216, (long long)NL_ * DFF_ * D_);

  auto gemm = [&](int outmode, int relu,
                  const _Float16* A, int lda, long long sA,
                  const _Float16* Bm, int ldb, long long sB,
                  void* C, int ldc, long long sC,
                  const float* bias, const float* res, long long sR,
                  int M, int N, int Kd, int batch) {
    dim3 grid(N / 64, (M + 255) / 256, batch);
    dim3 blk(128);
    if (outmode == 0)
      gemm16<0, false><<<grid, blk, 0, stream>>>(A, lda, sA, Bm, ldb, sB, C, ldc, sC,
                                                 bias, res, sR, M, N, Kd);
    else if (outmode == 1 && !relu)
      gemm16<1, false><<<grid, blk, 0, stream>>>(A, lda, sA, Bm, ldb, sB, C, ldc, sC,
                                                 bias, res, sR, M, N, Kd);
    else if (outmode == 1 && relu)
      gemm16<1, true><<<grid, blk, 0, stream>>>(A, lda, sA, Bm, ldb, sB, C, ldc, sC,
                                                bias, res, sR, M, N, Kd);
    else
      gemm16<2, false><<<grid, blk, 0, stream>>>(A, lda, sA, Bm, ldb, sB, C, ldc, sC,
                                                 bias, res, sR, M, N, Kd);
  };

  const float scale = 0.0625f;  // 1/sqrt(DH_)

  embed_kernel<<<BL_, 256, 0, stream>>>(cache, seq, item, cpos, spos, seg, h);

  for (int l = 0; l < NL_; ++l) {
    const _Float16* wq_s_l = wq_s16 + (size_t)l * D_ * DH_;
    const _Float16* wk_s_l = wk_s16 + (size_t)l * D_ * DH_;
    const _Float16* wv_s_l = wv_s16 + (size_t)l * D_ * DH_;
    const _Float16* wq_c_l = wq_c16 + (size_t)l * D_ * DH_;
    const _Float16* wk_c_l = wk_c16 + (size_t)l * D_ * DH_;
    const _Float16* wv_c_l = wv_c16 + (size_t)l * D_ * DH_;

    // hn16 = LN1(h)
    ln16_kernel<<<BL_, 256, 0, stream>>>(h, ln1_g + l * D_, ln1_b + l * D_, hn16, D_);

    // projections (q/v natural f16, k stored transposed f16)
    gemm(1, 0, hn16, D_, 0, wq_s_l, DH_, 0, qa16, DH_, 0,
         nullptr, nullptr, 0, BL_, DH_, D_, 1);
    gemm(2, 0, hn16 + (long long)K_ * D_, D_, (long long)L_ * D_, wk_s_l, DH_, 0,
         kaT, W_, (long long)DH_ * W_, nullptr, nullptr, 0, W_, DH_, D_, B_);
    gemm(1, 0, hn16 + (long long)K_ * D_, D_, (long long)L_ * D_, wv_s_l, DH_, 0,
         va16, DH_, (long long)W_ * DH_, nullptr, nullptr, 0, W_, DH_, D_, B_);
    gemm(1, 0, hn16, D_, 0, wq_c_l, DH_, 0, qb16, DH_, 0,
         nullptr, nullptr, 0, BL_, DH_, D_, 1);
    gemm(2, 0, hn16, D_, (long long)L_ * D_, wk_c_l, DH_, 0,
         kbT, K_, (long long)DH_ * K_, nullptr, nullptr, 0, K_, DH_, D_, B_);
    gemm(1, 0, hn16, D_, (long long)L_ * D_, wv_c_l, DH_, 0,
         vb16, DH_, (long long)K_ * DH_, nullptr, nullptr, 0, K_, DH_, D_, B_);

    // seq attention: scores = qa @ kaT (NN), softmax, ctx -> at16[:, :256]
    gemm(1, 0, qa16, DH_, (long long)L_ * DH_, kaT, W_, (long long)DH_ * W_,
         sc16, W_, (long long)L_ * W_, nullptr, nullptr, 0, L_, W_, DH_, B_);
    softmax16_kernel<<<BL_, 256, 0, stream>>>(sc16, W_, scale, L_, K_, 1);
    gemm(1, 0, sc16, W_, (long long)L_ * W_, va16, DH_, (long long)W_ * DH_,
         at16, D_, (long long)L_ * D_, nullptr, nullptr, 0, L_, DH_, W_, B_);

    // cache attention: scores_b = qb @ kbT (NN), softmax, ctx -> at16[:, 256:]
    gemm(1, 0, qb16, DH_, (long long)L_ * DH_, kbT, K_, (long long)DH_ * K_,
         scb16, K_, (long long)L_ * K_, nullptr, nullptr, 0, L_, K_, DH_, B_);
    softmax16_kernel<<<BL_, 256, 0, stream>>>(scb16, K_, scale, L_, K_, 0);
    gemm(1, 0, scb16, K_, (long long)L_ * K_, vb16, DH_, (long long)K_ * DH_,
         (void*)(at16 + DH_), D_, (long long)L_ * D_, nullptr, nullptr, 0,
         L_, DH_, K_, B_);

    // h = h + concat(out_a,out_b) @ w_out + b_out   (f32 accum + residual)
    gemm(0, 0, at16, D_, 0, w_out16 + (size_t)l * D_ * D_, D_, 0,
         h, D_, 0, b_out + l * D_, h, 0, BL_, D_, D_, 1);

    // FFN: h = h + relu(LN2(h) @ w_ff1 + b_ff1) @ w_ff2 + b_ff2
    ln16_kernel<<<BL_, 256, 0, stream>>>(h, ln2_g + l * D_, ln2_b + l * D_, hn16, D_);
    gemm(1, 1, hn16, D_, 0, w_ff116 + (size_t)l * D_ * DFF_, DFF_, 0,
         sc16, DFF_, 0, b_ff1 + l * DFF_, nullptr, 0, BL_, DFF_, D_, 1);
    gemm(0, 0, sc16, DFF_, 0, w_ff216 + (size_t)l * DFF_ * D_, D_, 0,
         h, D_, 0, b_ff2 + l * D_, h, 0, BL_, D_, DFF_, 1);
  }

  // final LN + eviction head over the K cache rows
  ln16_kernel<<<BL_, 256, 0, stream>>>(h, fin_g, fin_b, hn16, D_);
  logits_kernel<<<B_ * K_, 256, 0, stream>>>(hn16, w_ev, b_ev, (float*)d_out);
}
